// Encoder_90460601188434
// MI455X (gfx1250) — matrix operand order repaired and checked by reading, not compiled, not run on hardware
//
#include <hip/hip_runtime.h>
#include <hip/hip_bf16.h>

typedef __attribute__((ext_vector_type(16))) __bf16 v16bf;
typedef __attribute__((ext_vector_type(8)))  float  v8f;
typedef __attribute__((ext_vector_type(4)))  unsigned int v4u;

#define BATCH 2
#define HH 480
#define WW 640
#define HWSZ (HH*WW)
#define OUTC 160
#define HO 120
#define WO 160
#define NE 4
#define GIN_SZ (32*HO*WO)   /* 614400 */

#define FLAG_RELU  1
#define FLAG_ACCUM 2
#define FLAG_SCALE 4
#define FLAG_EXITZ 8

static __device__ __forceinline__ __bf16 f2bf(float f) {
  unsigned u = __builtin_bit_cast(unsigned, f);
  u += 0x7FFFu + ((u >> 16) & 1u);           // round-to-nearest-even
  unsigned short h = (unsigned short)(u >> 16);
  return __builtin_bit_cast(__bf16, h);
}

union FragU { v4u u[2]; v16bf v; };

// Generic implicit-GEMM 3x3 same-conv via V_WMMA_F32_16X16X32_BF16.
// A = weights (M=16 outch x K=32 ch-chunk), B = activations (K x N=16 pixels).
// 9 kernel taps * (CPAD/32) channel chunks accumulate into one f32 C tile.
// Block = 128 threads (4 wave32), each wave owns 16 consecutive x pixels;
// block covers 64 pixels of one output row, all 16 out channels.
template<int CPAD>
__global__ __launch_bounds__(128)
void conv3x3_wmma(const float* __restrict__ src, int srcCtot,
                  int ci0, int ci1, int csplit,
                  const float* __restrict__ W0, const float* __restrict__ W1,
                  int wCin, int wsplit,
                  const float* __restrict__ bias,
                  int Cin, int Cout,
                  float* __restrict__ dst, int dstCtot, int co,
                  const float* __restrict__ att, int attMode,
                  const float* __restrict__ scale, int sOff, int sBstr,
                  int mode)
{
  const int tid  = threadIdx.x;
  const int lane = tid & 31;
  const int wv   = tid >> 5;
  const int X0   = blockIdx.x * 64;
  const int y    = blockIdx.y;
  const int b    = blockIdx.z;

  float sc = 1.0f;
  if (mode & (FLAG_SCALE | FLAG_EXITZ)) {
    sc = scale[sOff + b * sBstr];
    if ((mode & FLAG_EXITZ) && sc == 0.0f) return;   // uniform per block
  }

  __shared__ __align__(16) __bf16 sIn[3][66][CPAD];   // 3 rows + x halo
  __shared__ __align__(16) __bf16 sW [9][16][CPAD];   // [tap][outch][inch]

  // stage weights -> bf16 LDS (zero-padded in outch/inch)
  for (int idx = tid; idx < 9*16*CPAD; idx += 128) {
    int c = idx % CPAD;
    int o = (idx / CPAD) % 16;
    int p = idx / (CPAD*16);
    float v = 0.0f;
    if (o < Cout && c < Cin) {
      const float* Wt = W0; int cl = c;
      if (c >= wsplit) { Wt = W1; cl = c - wsplit; }
      v = Wt[(((size_t)o*wCin + cl)*3 + (p/3))*3 + (p%3)];
    }
    sW[p][o][c] = f2bf(v);
  }

  // stage activations -> bf16 LDS (zero border, channel remap, optional att gate)
  for (int idx = tid; idx < 3*66*CPAD; idx += 128) {
    int col = idx % 66;
    int t   = idx / 66;
    int c   = t % CPAD;
    int ky  = t / CPAD;
    int gy = y + ky - 1;
    int gx = X0 + col - 1;
    float v = 0.0f;
    if (c < Cin && gy >= 0 && gy < HH && gx >= 0 && gx < WW) {
      int sch = (c < csplit) ? (ci0 + c) : (ci1 + c - csplit);
      v = src[((size_t)b*srcCtot + sch)*HWSZ + (size_t)gy*WW + gx];
      if (attMode) {
        float a = att[(size_t)b*HWSZ + (size_t)gy*WW + gx];
        v *= (attMode == 1) ? a : (1.0f - a);
      }
    }
    sIn[ky][col][c] = f2bf(v);
  }

  __syncthreads();

  v8f acc = {0.f,0.f,0.f,0.f,0.f,0.f,0.f,0.f};
  const int m  = lane & 15;   // A row (outch) / B col (pixel)
  const int hi = lane >> 4;

  #pragma unroll
  for (int p = 0; p < 9; ++p) {
    const int ky = p / 3, kx = p % 3;
    const int col = wv*16 + m + kx;            // 0..65 incl. halo
    #pragma unroll
    for (int ck = 0; ck < CPAD/32; ++ck) {
      FragU fa, fb;
      // A frag (ISA 16-bit A layout): halves {base..base+7, base+16..base+23}
      const __bf16* pa = &sW[p][m][ck*32 + hi*8];
      fa.u[0] = *reinterpret_cast<const v4u*>(pa);
      fa.u[1] = *reinterpret_cast<const v4u*>(pa + 16);
      // B frag: same K order as A: halves {base..base+7, base+16..base+23}
      const __bf16* pb = &sIn[ky][col][ck*32 + hi*8];
      fb.u[0] = *reinterpret_cast<const v4u*>(pb);
      fb.u[1] = *reinterpret_cast<const v4u*>(pb + 16);
      acc = __builtin_amdgcn_wmma_f32_16x16x32_bf16(
              false, fa.v, false, fb.v, (short)0, acc, false, false);
    }
  }

  // C/D layout: VGPR j, lanes 0-15 -> outch j, lanes 16-31 -> outch j+8; N=pixel
  const int x = X0 + wv*16 + m;
  #pragma unroll
  for (int j = 0; j < 8; ++j) {
    int oc = j + hi*8;
    if (oc >= Cout) continue;
    float v = acc[j] + bias[oc];
    if (mode & FLAG_RELU)  v = fmaxf(v, 0.0f);
    if (mode & FLAG_SCALE) v *= sc;
    float* po = dst + ((size_t)b*dstCtot + co + oc)*HWSZ + (size_t)y*WW + x;
    if (mode & FLAG_ACCUM) v += *po;
    *po = v;
  }
}

// bilinear 0.25x (align_corners) of att-gated [x3r|x3i]; out [moe][b][32][120][160]
__global__ void down4_kernel(const float* __restrict__ enc,
                             const float* __restrict__ att,
                             float* __restrict__ ds)
{
  int idx = blockIdx.x*blockDim.x + threadIdx.x;
  const int total = 2*BATCH*GIN_SZ;
  if (idx >= total) return;
  int xo = idx % WO;
  int yo = (idx / WO) % HO;
  int c  = (idx / (WO*HO)) % 32;
  int b  = (idx / (WO*HO*32)) % BATCH;
  int moe = idx / (WO*HO*32*BATCH);
  int ch = (c < 16) ? (48 + c) : (112 + (c - 16));
  float fy = yo * (479.0f/119.0f);
  float fx = xo * (639.0f/159.0f);
  int y0 = (int)floorf(fy); int y1 = (y0+1 < HH) ? y0+1 : HH-1; float wy = fy - y0;
  int x0 = (int)floorf(fx); int x1 = (x0+1 < WW) ? x0+1 : WW-1; float wx = fx - x0;
  const float* p  = enc + ((size_t)b*OUTC + ch)*HWSZ;
  const float* pa = att + (size_t)b*HWSZ;
  auto val = [&](int yy, int xx) {
    float a = pa[yy*WW + xx];
    if (moe) a = 1.0f - a;
    return p[yy*WW + xx] * a;
  };
  float r0 = val(y0,x0)*(1.0f-wx) + val(y0,x1)*wx;
  float r1 = val(y1,x0)*(1.0f-wx) + val(y1,x1)*wx;
  ds[idx] = r0*(1.0f-wy) + r1*wy;
}

__global__ void zero_kernel(float* __restrict__ p, int n) {
  int i = blockIdx.x*blockDim.x + threadIdx.x;
  if (i < n) p[i] = 0.0f;
}

// clean/noise-lin matvec: acc[moe][b][{0..3:gate, 4..7:noise}] += gin . W
__global__ void gate_matvec(const float* __restrict__ ds,
                            const float* __restrict__ wg_fg, const float* __restrict__ wn_fg,
                            const float* __restrict__ wg_bg, const float* __restrict__ wn_bg,
                            float* __restrict__ acc)
{
  int mb  = blockIdx.y;           // 0..3
  int moe = mb >> 1, b = mb & 1;
  const float* gin = ds + ((size_t)(moe*BATCH + b))*GIN_SZ;
  const float* wg  = moe ? wg_bg : wg_fg;
  const float* wn  = moe ? wn_bg : wn_fg;
  float s[8] = {0,0,0,0,0,0,0,0};
  for (int k = blockIdx.x*256 + threadIdx.x; k < GIN_SZ; k += gridDim.x*256) {
    float g = gin[k];
    const float* wgr = wg + 4*(size_t)k;
    const float* wnr = wn + 4*(size_t)k;
    #pragma unroll
    for (int e = 0; e < 4; ++e) { s[e] += g*wgr[e]; s[4+e] += g*wnr[e]; }
  }
  __shared__ float red[256];
  for (int i = 0; i < 8; ++i) {
    red[threadIdx.x] = s[i];
    __syncthreads();
    for (int off = 128; off > 0; off >>= 1) {
      if (threadIdx.x < off) red[threadIdx.x] += red[threadIdx.x + off];
      __syncthreads();
    }
    if (threadIdx.x == 0) atomicAdd(&acc[mb*8 + i], red[0]);
    __syncthreads();
  }
}

// noisy top-k gating + load-balance aux (tiny: E=4, B=2, serial on 1 thread)
__global__ void gate_finalize(const float* __restrict__ acc,
                              const float* __restrict__ noise_fg,
                              const float* __restrict__ noise_bg,
                              float* __restrict__ gates,
                              float* __restrict__ aux_out)
{
  if (threadIdx.x != 0 || blockIdx.x != 0) return;
  float aux_total = 0.0f;
  for (int moe = 0; moe < 2; ++moe) {
    const float* noise = moe ? noise_bg : noise_fg;
    float g[BATCH][NE], prob[BATCH][NE];
    for (int b = 0; b < BATCH; ++b) {
      const float* a = acc + (moe*BATCH + b)*8;
      float clean[NE], stdv[NE], noisy[NE];
      for (int e = 0; e < NE; ++e) {
        clean[e] = a[e];
        float sp  = a[4+e];
        float spv = (sp > 20.0f) ? sp : log1pf(expf(sp));  // softplus
        stdv[e]  = spv + 1e-2f;
        noisy[e] = clean[e] + noise[b*NE + e]*stdv[e];
      }
      int ord[NE] = {0,1,2,3};
      for (int i = 0; i < NE; ++i)
        for (int j = i+1; j < NE; ++j)
          if (noisy[ord[j]] > noisy[ord[i]]) { int t = ord[i]; ord[i] = ord[j]; ord[j] = t; }
      float tv0 = noisy[ord[0]], tv1 = noisy[ord[1]], tv2 = noisy[ord[2]];
      float mx = fmaxf(tv0, tv1);
      float e0 = expf(tv0 - mx), e1 = expf(tv1 - mx);
      float den = e0 + e1;
      for (int e = 0; e < NE; ++e) g[b][e] = 0.0f;
      g[b][ord[0]] = e0/den;
      g[b][ord[1]] = e1/den;
      for (int e = 0; e < NE; ++e) {
        bool isin = noisy[e] > tv2;
        float z = (clean[e] - (isin ? tv2 : tv1)) / stdv[e];
        prob[b][e] = 0.5f*(1.0f + erff(z*0.70710678118f)); // Phi(z)
      }
      for (int e = 0; e < NE; ++e) gates[(moe*BATCH + b)*NE + e] = g[b][e];
    }
    float imp[NE], ld[NE], mi = 0.0f, ml = 0.0f;
    for (int e = 0; e < NE; ++e) {
      imp[e] = 0.0f; ld[e] = 0.0f;
      for (int b = 0; b < BATCH; ++b) { imp[e] += g[b][e]; ld[e] += prob[b][e]; }
      mi += imp[e]; ml += ld[e];
    }
    mi /= NE; ml /= NE;
    float vi = 0.0f, vl = 0.0f;
    for (int e = 0; e < NE; ++e) {
      vi += (imp[e]-mi)*(imp[e]-mi);
      vl += (ld[e]-ml)*(ld[e]-ml);
    }
    vi /= (NE-1); vl /= (NE-1);
    aux_total += (vi/(mi*mi + 1e-10f) + vl/(ml*ml + 1e-10f)) * 0.01f;
  }
  aux_out[0] = aux_total;
}

extern "C" void kernel_launch(void* const* d_in, const int* in_sizes, int n_in,
                              void* d_out, int out_size, void* d_ws, size_t ws_size,
                              hipStream_t stream)
{
  (void)in_sizes; (void)n_in; (void)out_size; (void)ws_size;
  // setup_inputs() depth-first dict order
  const float* x_r    = (const float*)d_in[0];
  const float* x_i    = (const float*)d_in[1];
  const float* att    = (const float*)d_in[2];
  const float* c1r_w  = (const float*)d_in[3];
  const float* c1r_b  = (const float*)d_in[4];
  const float* c1i_w  = (const float*)d_in[5];
  const float* c1i_b  = (const float*)d_in[6];
  const float* d1r_w  = (const float*)d_in[7];
  const float* d1r_b  = (const float*)d_in[8];
  const float* d1i_w  = (const float*)d_in[9];
  const float* d1i_b  = (const float*)d_in[10];
  const float* d2r_w  = (const float*)d_in[11];
  const float* d2r_b  = (const float*)d_in[12];
  const float* d2i_w  = (const float*)d_in[13];
  const float* d2i_b  = (const float*)d_in[14];
  const float* d3r_w  = (const float*)d_in[15];
  const float* d3r_b  = (const float*)d_in[16];
  const float* d3i_w  = (const float*)d_in[17];
  const float* d3i_b  = (const float*)d_in[18];
  const float* wg_fg  = (const float*)d_in[19];
  const float* wn_fg  = (const float*)d_in[20];
  const float* wr_fg  = (const float*)d_in[21];
  const float* wi_fg  = (const float*)d_in[22];
  const float* bh_fg  = (const float*)d_in[23];
  const float* wo_fg  = (const float*)d_in[24];
  const float* bo_fg  = (const float*)d_in[25];
  const float* wg_bg  = (const float*)d_in[26];
  const float* wn_bg  = (const float*)d_in[27];
  const float* wr_bg  = (const float*)d_in[28];
  const float* wi_bg  = (const float*)d_in[29];
  const float* bh_bg  = (const float*)d_in[30];
  const float* wo_bg  = (const float*)d_in[31];
  const float* bo_bg  = (const float*)d_in[32];
  const float* noise_fg = (const float*)d_in[33];
  const float* noise_bg = (const float*)d_in[34];

  float* out = (float*)d_out;
  float* ws    = (float*)d_ws;
  float* ds    = ws;                              // 2*BATCH*GIN_SZ
  float* gacc  = ds + (size_t)2*BATCH*GIN_SZ;     // 32
  float* gates = gacc + 32;                       // 16
  float* hbuf  = gates + 16;                      // BATCH*8*HWSZ

  dim3 cgrid(WW/64, HH, BATCH);
  dim3 cblk(128);

  // front 1->16 convs + relu (out ch 0-15 / 64-79)
  conv3x3_wmma<32><<<cgrid,cblk,0,stream>>>(x_r,1, 0,0,999, c1r_w,c1r_w,1,999, c1r_b, 1,16,
                                            out,OUTC,0,  nullptr,0, nullptr,0,0, FLAG_RELU);
  conv3x3_wmma<32><<<cgrid,cblk,0,stream>>>(x_i,1, 0,0,999, c1i_w,c1i_w,1,999, c1i_b, 1,16,
                                            out,OUTC,64, nullptr,0, nullptr,0,0, FLAG_RELU);
  // dense block d1: 16->16 (out ch 16-31 / 80-95)
  conv3x3_wmma<32><<<cgrid,cblk,0,stream>>>(out,OUTC, 0,0,999,  d1r_w,d1r_w,16,999, d1r_b, 16,16,
                                            out,OUTC,16, nullptr,0, nullptr,0,0, 0);
  conv3x3_wmma<32><<<cgrid,cblk,0,stream>>>(out,OUTC, 64,0,999, d1i_w,d1i_w,16,999, d1i_b, 16,16,
                                            out,OUTC,80, nullptr,0, nullptr,0,0, 0);
  // d2: 32->16 (out ch 32-47 / 96-111)
  conv3x3_wmma<32><<<cgrid,cblk,0,stream>>>(out,OUTC, 0,0,999,  d2r_w,d2r_w,32,999, d2r_b, 32,16,
                                            out,OUTC,32, nullptr,0, nullptr,0,0, 0);
  conv3x3_wmma<32><<<cgrid,cblk,0,stream>>>(out,OUTC, 64,0,999, d2i_w,d2i_w,32,999, d2i_b, 32,16,
                                            out,OUTC,96, nullptr,0, nullptr,0,0, 0);
  // d3: 48->16 (out ch 48-63 / 112-127)
  conv3x3_wmma<64><<<cgrid,cblk,0,stream>>>(out,OUTC, 0,0,999,  d3r_w,d3r_w,48,999, d3r_b, 48,16,
                                            out,OUTC,48,  nullptr,0, nullptr,0,0, 0);
  conv3x3_wmma<64><<<cgrid,cblk,0,stream>>>(out,OUTC, 64,0,999, d3i_w,d3i_w,48,999, d3i_b, 48,16,
                                            out,OUTC,112, nullptr,0, nullptr,0,0, 0);

  // gating: downsample -> matvec -> finalize
  {
    int tot = 2*BATCH*GIN_SZ;
    down4_kernel<<<(tot+255)/256,256,0,stream>>>(out, att, ds);
    zero_kernel<<<1,32,0,stream>>>(gacc, 32);
    gate_matvec<<<dim3(240,4),256,0,stream>>>(ds, wg_fg, wn_fg, wg_bg, wn_bg, gacc);
    gate_finalize<<<1,1,0,stream>>>(gacc, noise_fg, noise_bg, gates,
                                    out + (size_t)BATCH*OUTC*HWSZ);
  }

  // MoE experts: h = relu(conv([x3r*att | x3i*att], [w_r|w_i]) + b_h); y += g*(conv(h,w_o)+b_o)
  for (int moe = 0; moe < 2; ++moe) {
    const float* w_r = moe ? wr_bg : wr_fg;
    const float* w_i = moe ? wi_bg : wi_fg;
    const float* b_h = moe ? bh_bg : bh_fg;
    const float* w_o = moe ? wo_bg : wo_fg;
    const float* b_o = moe ? bo_bg : bo_fg;
    for (int e = 0; e < NE; ++e) {
      int m1 = FLAG_RELU | (e ? FLAG_EXITZ : 0);
      conv3x3_wmma<32><<<cgrid,cblk,0,stream>>>(
          out, OUTC, 48, 112, 16,
          w_r + (size_t)e*1152, w_i + (size_t)e*1152, 16, 16,
          b_h + e*8, 32, 8,
          hbuf, 8, 0, att, moe+1, gates, moe*8 + e, 4, m1);
      int m2 = FLAG_SCALE | (e ? (FLAG_ACCUM|FLAG_EXITZ) : 0);
      conv3x3_wmma<32><<<cgrid,cblk,0,stream>>>(
          hbuf, 8, 0, 0, 999,
          w_o + (size_t)e*1152, w_o + (size_t)e*1152, 8, 999,
          b_o + e*16, 8, 16,
          out, OUTC, 128 + 16*moe, nullptr, 0, gates, moe*8 + e, 4, m2);
    }
  }
}
